// StackedAttentionLSTM_9740985828129
// MI455X (gfx1250) — compile-verified, matrix operand order checked by reading, not run on hardware
//
#include <hip/hip_runtime.h>
#include <cstdint>
#include <cstddef>

typedef __bf16 bf16;
typedef __attribute__((ext_vector_type(16))) __bf16 v16bf;
typedef __attribute__((ext_vector_type(8)))  __bf16 v8bf;
typedef __attribute__((ext_vector_type(8)))  float   v8f;

namespace {
constexpr int kB = 64, kT = 512, kS = 512, kD = 1024, kH = 1024;
constexpr int kG = 4 * kH;   // 4096 gate columns
constexpr int GK = 32;       // K-slice per WMMA step
}

__device__ __forceinline__ bf16 f2bf(float f) {
  union { float f; uint32_t u; } v; v.f = f;
  uint32_t u = v.u + 0x7fffu + ((v.u >> 16) & 1u);   // RNE
  unsigned short h = (unsigned short)(u >> 16);
  return __builtin_bit_cast(bf16, h);
}

// 16x32 bf16 WMMA fragment from a row-major tile (global or LDS).
// lane l: row = l%16, half = l/16; j=0..7 -> k = half*8 + j,
// j=8..15 -> k = 16 + half*8 + (j-8): two contiguous 16-byte loads.
__device__ __forceinline__ v16bf load_frag(const bf16* __restrict__ p, int ld, int lane) {
  const int r = lane & 15, half = lane >> 4;
  const v8bf lo = *reinterpret_cast<const v8bf*>(p + (size_t)r * ld + half * 8);
  const v8bf hi = *reinterpret_cast<const v8bf*>(p + (size_t)r * ld + 16 + half * 8);
  return __builtin_shufflevector(lo, hi, 0,1,2,3,4,5,6,7,8,9,10,11,12,13,14,15);
}

__device__ __forceinline__ v8f wmma_bf16(v16bf a, v16bf b, v8f c) {
  return __builtin_amdgcn_wmma_f32_16x16x32_bf16(false, a, false, b, (short)0, c, false, false);
}

// CDNA5 async global->LDS copy, 16B per lane, tracked with ASYNCcnt.
__device__ __forceinline__ void async_copy_b128(void* lds_dst, const void* gsrc) {
  const uint32_t lds_off = (uint32_t)(uintptr_t)lds_dst;      // flat low 32 == LDS offset
  asm volatile("global_load_async_to_lds_b128 %0, %1, off"
               :: "v"(lds_off), "v"((uint64_t)(uintptr_t)gsrc)
               : "memory");
}
__device__ __forceinline__ void wait_async0() {
  asm volatile("s_wait_asynccnt 0x0" ::: "memory");
}

// ---------------------------------------------------------------- utilities
__global__ void cvt_kernel(const float* __restrict__ in, bf16* __restrict__ out, size_t n) {
  size_t i = (size_t)blockIdx.x * blockDim.x + threadIdx.x;
  if (i < n) out[i] = f2bf(in[i]);
}

__global__ void addvec_kernel(const float* __restrict__ a, const float* __restrict__ b,
                              float* __restrict__ o, int n) {
  int i = blockIdx.x * blockDim.x + threadIdx.x;
  if (i < n) o[i] = a[i] + b[i];
}

__global__ void copy_rows_kernel(const float* __restrict__ src, size_t srow,
                                 float* __restrict__ dst, int rows, int cols) {
  int i = blockIdx.x * blockDim.x + threadIdx.x;
  if (i < rows * cols) {
    int r = i / cols, c = i % cols;
    dst[(size_t)r * cols + c] = src[(size_t)r * srow + c];
  }
}

// ------------------------------------------------- big parallel input GEMM
// C[M,N] = A[M,K] @ W[N,K]^T + bias[n].
// Block = 256 threads (8 wave32) -> 128(M) x 64(N) tile. Per k-step, the
// 64x32 W strip (shared by all 8 waves) and the 128x32 A block are staged
// into LDS with async b128 copies (double buffered, ASYNCcnt), then each
// wave ds_loads its fragments. All 4 B fragments are fetched before the
// WMMA chain so the dscnt waits can overlap WMMA issue.
__global__ void gemm_bf16_lds(const bf16* __restrict__ A, const bf16* __restrict__ W,
                              const float* __restrict__ bias, float* __restrict__ C,
                              int M, int K, int N) {
  __shared__ __align__(16) bf16 sW[2][64 * GK];    // 4 KB per buffer
  __shared__ __align__(16) bf16 sA[2][128 * GK];   // 8 KB per buffer
  const int tid  = threadIdx.x;
  const int lane = tid & 31, wave = tid >> 5;
  const int n0   = blockIdx.x * 64;
  const bf16* Ablk = A + (size_t)blockIdx.y * 128 * K;

  // 768 16-byte chunks per stage: 256 W chunks + 512 A chunks, 3 per thread.
  auto stage = [&](int buf, int k0) {
    {
      const int row = tid >> 2, piece = tid & 3;                 // W: 64 rows x 4 pieces
      async_copy_b128(&sW[buf][row * GK + piece * 8],
                      W + (size_t)(n0 + row) * K + k0 + piece * 8);
    }
#pragma unroll
    for (int rr = 0; rr < 2; ++rr) {                             // A: 128 rows x 4 pieces
      const int c = rr * 256 + tid;
      const int row = c >> 2, piece = c & 3;
      async_copy_b128(&sA[buf][row * GK + piece * 8],
                      Ablk + (size_t)row * K + k0 + piece * 8);
    }
  };

  v8f acc[4] = {};
  stage(0, 0);
  const int nsteps = K / GK;
  for (int s = 0; s < nsteps; ++s) {
    wait_async0();            // this thread's stage for buffer (s&1) has landed
    __syncthreads();          // publish all waves' copies; fences reuse of buf ((s+1)&1)
    if (s + 1 < nsteps) stage((s + 1) & 1, (s + 1) * GK);   // overlap with compute
    const int buf = s & 1;
    // fetch A fragment and all four B fragments first, then the WMMA chain
    v16bf a = load_frag(&sA[buf][wave * 16 * GK], GK, lane);
    v16bf b0 = load_frag(&sW[buf][0 * 16 * GK], GK, lane);
    v16bf b1 = load_frag(&sW[buf][1 * 16 * GK], GK, lane);
    v16bf b2 = load_frag(&sW[buf][2 * 16 * GK], GK, lane);
    v16bf b3 = load_frag(&sW[buf][3 * 16 * GK], GK, lane);
    acc[0] = wmma_bf16(a, b0, acc[0]);
    acc[1] = wmma_bf16(a, b1, acc[1]);
    acc[2] = wmma_bf16(a, b2, acc[2]);
    acc[3] = wmma_bf16(a, b3, acc[3]);
  }
  const int nc = lane & 15, half = lane >> 4;
#pragma unroll
  for (int j = 0; j < 4; ++j) {
    const int n = n0 + j * 16 + nc;
    const float bv = bias[n];
#pragma unroll
    for (int r = 0; r < 8; ++r) {
      const int m = blockIdx.y * 128 + wave * 16 + r + 8 * half;
      C[(size_t)m * N + n] = acc[j][r] + bv;
    }
  }
}

// ------------------------------------------- serial step 1: gates + LSTM cell
// gates = Xi_t + h_prev @ Wh^T ; one wave owns a 16(batch) x 16(hidden) region
// and accumulates all four gates (i,f,g,o) so the cell update fuses in-register.
// Weights are L2-resident; direct loads keep the serial chain barrier-free.
// All 8 B-fragment loads issue as one clause before the 4-WMMA chain.
__global__ void lstm_recur_kernel(const bf16* __restrict__ hprev,   // [B,H] bf16
                                  const bf16* __restrict__ Wh,      // [G,H] bf16
                                  const float* __restrict__ Xi_t,   // t-offset base
                                  size_t xi_mstride,                // stride per batch row
                                  float* __restrict__ cbuf,         // [B,H]
                                  float* __restrict__ hyf,          // [B,H] f32 (attention)
                                  bf16*  __restrict__ concatb) {    // [B,2H], hy -> +H half
  const int lane = threadIdx.x & 31;
  const int wave = threadIdx.x >> 5;
  const int gw   = blockIdx.x * (blockDim.x >> 5) + wave;  // 0..255
  const int mt   = gw & 3;                                 // batch tile
  const int ht   = gw >> 2;                                // hidden tile 0..63
  v8f acc[4] = {};
  const bf16* Arow = hprev + (size_t)mt * 16 * kH;
  for (int k0 = 0; k0 < kH; k0 += GK) {
    v16bf a  = load_frag(Arow + k0, kH, lane);
    v16bf b0 = load_frag(Wh + (size_t)(0 * kH + ht * 16) * kH + k0, kH, lane);
    v16bf b1 = load_frag(Wh + (size_t)(1 * kH + ht * 16) * kH + k0, kH, lane);
    v16bf b2 = load_frag(Wh + (size_t)(2 * kH + ht * 16) * kH + k0, kH, lane);
    v16bf b3 = load_frag(Wh + (size_t)(3 * kH + ht * 16) * kH + k0, kH, lane);
    acc[0] = wmma_bf16(a, b0, acc[0]);
    acc[1] = wmma_bf16(a, b1, acc[1]);
    acc[2] = wmma_bf16(a, b2, acc[2]);
    acc[3] = wmma_bf16(a, b3, acc[3]);
  }
  const int nc = lane & 15, half = lane >> 4;
  const int hcol = ht * 16 + nc;
#pragma unroll
  for (int r = 0; r < 8; ++r) {
    const int m = mt * 16 + r + 8 * half;
    const float* xr = Xi_t + (size_t)m * xi_mstride;
    float gi = acc[0][r] + xr[hcol];
    float gf = acc[1][r] + xr[kH + hcol];
    float gg = acc[2][r] + xr[2 * kH + hcol];
    float go = acc[3][r] + xr[3 * kH + hcol];
    gi = 1.f / (1.f + __expf(-gi));
    gf = 1.f / (1.f + __expf(-gf));
    gg = tanhf(gg);
    go = 1.f / (1.f + __expf(-go));
    const size_t ci = (size_t)m * kH + hcol;
    const float cy = gf * cbuf[ci] + gi * gg;
    cbuf[ci] = cy;
    const float hy = go * tanhf(cy);
    hyf[ci] = hy;
    concatb[(size_t)m * (2 * kH) + kH + hcol] = f2bf(hy);
  }
}

// ------------------------------------------- serial step 2: dot attention
// One block per batch row b. Batched GEMV (each b has its own ctx) -> VALU path.
__global__ void attn_kernel(const float* __restrict__ ctx,   // [S,B,H] as given
                            const float* __restrict__ hyf,   // [B,H]
                            bf16* __restrict__ concatb) {    // wctx -> first half
  __shared__ float s_attn[kS];
  __shared__ float s_red[512];
  const int b = blockIdx.x;
  const int tid = threadIdx.x;              // 0..511 (16 wave32s)
  const int lane = tid & 31, wave = tid >> 5;
  // scores[s] = hy . ctx[s,b,:]
  const float* hr = hyf + (size_t)b * kH;
  for (int i = 0; i < kS / 16; ++i) {
    const int s = i * 16 + wave;
    const float* cr = ctx + ((size_t)s * kB + b) * kH;
    float acc = 0.f;
    for (int d = lane; d < kH; d += 32) acc += hr[d] * cr[d];
    for (int off = 16; off; off >>= 1) acc += __shfl_xor(acc, off, 32);
    if (lane == 0) s_attn[s] = acc;
  }
  __syncthreads();
  // softmax over S == blockDim
  const float v = s_attn[tid];
  s_red[tid] = v;
  __syncthreads();
  for (int off = 256; off; off >>= 1) {
    if (tid < off) s_red[tid] = fmaxf(s_red[tid], s_red[tid + off]);
    __syncthreads();
  }
  const float mx = s_red[0];
  __syncthreads();
  const float e = __expf(v - mx);
  s_red[tid] = e;
  __syncthreads();
  for (int off = 256; off; off >>= 1) {
    if (tid < off) s_red[tid] += s_red[tid + off];
    __syncthreads();
  }
  const float inv = 1.f / s_red[0];
  __syncthreads();
  s_attn[tid] = e * inv;
  __syncthreads();
  // wctx[d] = sum_s attn[s] * ctx[s,b,d]
  for (int d = tid; d < kH; d += 512) {
    float acc = 0.f;
    for (int s = 0; s < kS; ++s)
      acc += s_attn[s] * ctx[((size_t)s * kB + b) * kH + d];
    concatb[(size_t)b * (2 * kH) + d] = f2bf(acc);
  }
}

// ------------------------------------------- serial step 3: output projection
// h_tilde = tanh([wctx|hy] @ Wout^T); writes f32 output + bf16 recurrent state.
__global__ void outproj_kernel(const bf16* __restrict__ concatb, // [B,2H]
                               const bf16* __restrict__ Wout,    // [H,2H]
                               float* __restrict__ outbase,      // t-offset base
                               size_t out_mstride,
                               bf16* __restrict__ htlb) {        // [B,H]
  const int lane = threadIdx.x & 31;
  const int wave = threadIdx.x >> 5;
  const int gw   = blockIdx.x * (blockDim.x >> 5) + wave;  // 0..255
  const int mt   = gw & 3;
  const int nt   = gw >> 2;                                // 0..63
  v8f acc = {};
  const bf16* Arow = concatb + (size_t)mt * 16 * (2 * kH);
  const bf16* Wrow = Wout + (size_t)nt * 16 * (2 * kH);
  for (int k0 = 0; k0 < 2 * kH; k0 += GK) {
    v16bf a = load_frag(Arow + k0, 2 * kH, lane);
    v16bf b = load_frag(Wrow + k0, 2 * kH, lane);
    acc = wmma_bf16(a, b, acc);
  }
  const int nc = lane & 15, half = lane >> 4;
  const int n = nt * 16 + nc;
#pragma unroll
  for (int r = 0; r < 8; ++r) {
    const int m = mt * 16 + r + 8 * half;
    const float v = tanhf(acc[r]);
    outbase[(size_t)m * out_mstride + n] = v;
    htlb[(size_t)m * kH + n] = f2bf(v);
  }
}

// ---------------------------------------------------------------------------
extern "C" void kernel_launch(void* const* d_in, const int* in_sizes, int n_in,
                              void* d_out, int out_size, void* d_ws, size_t ws_size,
                              hipStream_t stream) {
  const float* x     = (const float*)d_in[0];
  const float* h0    = (const float*)d_in[1];
  const float* c0    = (const float*)d_in[2];
  const float* ctx   = (const float*)d_in[3];
  const float* Wi0   = (const float*)d_in[4];
  const float* bi0   = (const float*)d_in[5];
  const float* Wh0   = (const float*)d_in[6];
  const float* bh0   = (const float*)d_in[7];
  const float* Wout0 = (const float*)d_in[8];
  const float* Wi1   = (const float*)d_in[9];
  const float* bi1   = (const float*)d_in[10];
  const float* Wh1   = (const float*)d_in[11];
  const float* bh1   = (const float*)d_in[12];
  const float* Wout1 = (const float*)d_in[13];
  float* out = (float*)d_out;
  (void)in_sizes; (void)n_in; (void)out_size; (void)ws_size;

  // --- workspace carve-up (256B aligned) ---
  uint8_t* wsb = (uint8_t*)d_ws;
  size_t off = 0;
  auto carve = [&](size_t bytes) {
    void* p = wsb + off;
    off = (off + bytes + 255) & ~(size_t)255;
    return p;
  };
  bf16*  wi0B    = (bf16*)carve((size_t)kG * kD * 2);
  bf16*  wh0B    = (bf16*)carve((size_t)kG * kH * 2);
  bf16*  wo0B    = (bf16*)carve((size_t)kH * 2 * kH * 2);
  bf16*  wi1B    = (bf16*)carve((size_t)kG * kH * 2);
  bf16*  wh1B    = (bf16*)carve((size_t)kG * kH * 2);
  bf16*  wo1B    = (bf16*)carve((size_t)kH * 2 * kH * 2);
  bf16*  xbf     = (bf16*)carve((size_t)kB * kT * kD * 2);
  float* Xi      = (float*)carve((size_t)kB * kT * kG * 4);
  float* out0    = (float*)carve((size_t)kB * kT * kH * 4);  // layer-0 outputs [T,B,H]
  float* bias0   = (float*)carve((size_t)kG * 4);
  float* bias1   = (float*)carve((size_t)kG * 4);
  float* cbuf    = (float*)carve((size_t)kB * kH * 4);
  float* hyf     = (float*)carve((size_t)kB * kH * 4);
  bf16*  concatb = (bf16*)carve((size_t)kB * 2 * kH * 2);
  bf16*  htlb    = (bf16*)carve((size_t)kB * kH * 2);

  const dim3 blk256(256), blk512(512);
  auto cvt = [&](const float* src, bf16* dst, size_t n) {
    cvt_kernel<<<dim3((unsigned)((n + 255) / 256)), blk256, 0, stream>>>(src, dst, n);
  };

  // weights -> bf16 once; fold (bi + bh) into the precomputed Xi bias
  cvt(Wi0, wi0B, (size_t)kG * kD);
  cvt(Wh0, wh0B, (size_t)kG * kH);
  cvt(Wout0, wo0B, (size_t)kH * 2 * kH);
  cvt(Wi1, wi1B, (size_t)kG * kH);
  cvt(Wh1, wh1B, (size_t)kG * kH);
  cvt(Wout1, wo1B, (size_t)kH * 2 * kH);
  addvec_kernel<<<dim3(kG / 256), blk256, 0, stream>>>(bi0, bh0, bias0, kG);
  addvec_kernel<<<dim3(kG / 256), blk256, 0, stream>>>(bi1, bh1, bias1, kG);

  const size_t OUT = (size_t)kB * kT * kH;     // d_out: out | h1(2,B,H) | c1(2,B,H)
  const dim3 gemm_grid(kG / 64, (kB * kT) / 128);
  const unsigned bh_blocks = (unsigned)((kB * kH + 255) / 256);

  // ================= layer 0 =================
  // Xi0[b*T+t, :] = x @ Wi0^T + (bi0+bh0)  (hoisted out of the serial loop)
  cvt(x, xbf, (size_t)kB * kT * kD);
  gemm_bf16_lds<<<gemm_grid, blk256, 0, stream>>>(xbf, wi0B, bias0, Xi, kB * kT, kD, kG);
  cvt(h0, htlb, (size_t)kB * kH);
  hipMemcpyAsync(cbuf, c0, (size_t)kB * kH * 4, hipMemcpyDeviceToDevice, stream);
  for (int t = 0; t < kT; ++t) {
    lstm_recur_kernel<<<dim3(32), blk256, 0, stream>>>(
        htlb, wh0B, Xi + (size_t)t * kG, (size_t)kT * kG, cbuf, hyf, concatb);
    attn_kernel<<<dim3(kB), blk512, 0, stream>>>(ctx, hyf, concatb);
    outproj_kernel<<<dim3(32), blk256, 0, stream>>>(
        concatb, wo0B, out0 + (size_t)t * kB * kH, (size_t)kH, htlb);
  }
  copy_rows_kernel<<<dim3(bh_blocks), blk256, 0, stream>>>(
      out0 + (size_t)(kT - 1) * kB * kH, (size_t)kH, out + OUT, kB, kH);            // h1[0]
  copy_rows_kernel<<<dim3(bh_blocks), blk256, 0, stream>>>(
      cbuf, (size_t)kH, out + OUT + 2 * (size_t)kB * kH, kB, kH);                   // c1[0]

  // ================= layer 1 =================
  // Xi1[t*B+b, :] = out0 @ Wi1^T + (bi1+bh1)
  cvt(out0, xbf, (size_t)kB * kT * kH);
  gemm_bf16_lds<<<gemm_grid, blk256, 0, stream>>>(xbf, wi1B, bias1, Xi, kB * kT, kH, kG);
  cvt(h0, htlb, (size_t)kB * kH);
  hipMemcpyAsync(cbuf, c0, (size_t)kB * kH * 4, hipMemcpyDeviceToDevice, stream);
  for (int t = 0; t < kT; ++t) {
    lstm_recur_kernel<<<dim3(32), blk256, 0, stream>>>(
        htlb, wh1B, Xi + (size_t)t * kB * kG, (size_t)kG, cbuf, hyf, concatb);
    attn_kernel<<<dim3(kB), blk512, 0, stream>>>(ctx, hyf, concatb);
    outproj_kernel<<<dim3(32), blk256, 0, stream>>>(
        concatb, wo1B, out + (size_t)t * kH, (size_t)kT * kH, htlb);   // straight to [B,T,H]
  }
  copy_rows_kernel<<<dim3(bh_blocks), blk256, 0, stream>>>(
      out + (size_t)(kT - 1) * kH, (size_t)kT * kH, out + OUT + (size_t)kB * kH, kB, kH); // h1[1]
  copy_rows_kernel<<<dim3(bh_blocks), blk256, 0, stream>>>(
      cbuf, (size_t)kH, out + OUT + 3 * (size_t)kB * kH, kB, kH);                         // c1[1]
}
